// SAGE_15985868276093
// MI455X (gfx1250) — matrix operand order repaired and checked by reading, not compiled, not run on hardware
//
#include <hip/hip_runtime.h>

typedef __attribute__((ext_vector_type(16))) __bf16 v16bf;
typedef __attribute__((ext_vector_type(8)))  __bf16 v8bf;
typedef __attribute__((ext_vector_type(4)))  __bf16 v4bf;
typedef __attribute__((ext_vector_type(8)))  float  v8f;

// ---------------- CSR construction ----------------

__global__ void k_zero_i32(int* __restrict__ p, int n) {
    int i = blockIdx.x * blockDim.x + threadIdx.x;
    if (i < n) p[i] = 0;
}

__global__ void k_degree(const int* __restrict__ edge, int E, int* __restrict__ deg) {
    int e = blockIdx.x * blockDim.x + threadIdx.x;
    if (e < E) atomicAdd(&deg[edge[e]], 1);   // edge row0 = dst
}

// single-block exclusive scan over N elements
__global__ void k_scan(const int* __restrict__ deg, int* __restrict__ rowptr, int n) {
    __shared__ int sums[1024];
    int t = threadIdx.x;
    int chunk = (n + 1023) >> 10;
    int lo = t * chunk;
    int hi = lo + chunk; if (hi > n) hi = n;
    int s = 0;
    for (int i = lo; i < hi; i++) s += deg[i];
    sums[t] = s;
    __syncthreads();
    for (int off = 1; off < 1024; off <<= 1) {
        int v = (t >= off) ? sums[t - off] : 0;
        __syncthreads();
        sums[t] += v;
        __syncthreads();
    }
    int run = (t == 0) ? 0 : sums[t - 1];
    for (int i = lo; i < hi; i++) { rowptr[i] = run; run += deg[i]; }
    if (t == 1023) rowptr[n] = sums[1023];
}

__global__ void k_fill(const int* __restrict__ edge, int E,
                       const int* __restrict__ rowptr, int* __restrict__ cnt,
                       int* __restrict__ col) {
    int e = blockIdx.x * blockDim.x + threadIdx.x;
    if (e < E) {
        int d = edge[e];
        int s = edge[E + e];
        int p = rowptr[d] + atomicAdd(&cnt[d], 1);
        col[p] = s;
    }
}

__global__ void k_deginv(const int* __restrict__ rowptr, float* __restrict__ dinv, int n) {
    int i = blockIdx.x * blockDim.x + threadIdx.x;
    if (i < n) {
        int d = rowptr[i + 1] - rowptr[i];
        dinv[i] = 1.0f / fmaxf((float)d, 1.0f);
    }
}

// ---------------- weight fragment pre-pack (once per layer, 16 KB) ----------------
// B-fragment map (contiguous k order used here, 16-bit B 32x16): b[i] = W[k0 + 16*half + i][n]
// packed index: Wp[((s*4 + nt)*32 + half*16 + (n&15))*16 + i]

__global__ void k_pack_w(const float* __restrict__ Wl, const float* __restrict__ Wr,
                         int dout, __bf16* __restrict__ Wp) {
    int idx = blockIdx.x * blockDim.x + threadIdx.x;   // 0..8191
    if (idx >= 128 * 64) return;
    int k = idx >> 6, n = idx & 63;
    float w = 0.f;
    if (n < dout) w = (k < 64) ? Wl[k * dout + n] : Wr[(k - 64) * dout + n];
    int s    = k >> 5;
    int kw   = k & 31;
    int half = kw >> 4;
    int i    = kw & 15;
    int lane = half * 16 + (n & 15);
    int nt   = n >> 4;
    Wp[(((s * 4 + nt) * 32) + lane) * 16 + i] = (__bf16)w;
}

// ---------------- mean aggregation (gather, one wave per node) ----------------

__global__ __launch_bounds__(256)
void k_gather(const float* __restrict__ feat, const int* __restrict__ rowptr,
              const int* __restrict__ col, const float* __restrict__ dinv,
              float* __restrict__ agg, int n) {
    int wave = threadIdx.x >> 5, lane = threadIdx.x & 31;
    int node = blockIdx.x * 8 + wave;
    if (node >= n) return;
    int beg = rowptr[node], end = rowptr[node + 1];
    const float2* f2 = (const float2*)feat;         // stride 32 float2 per node
    float ax = 0.f, ay = 0.f;
    for (int j0 = beg; j0 < end; j0 += 32) {
        int m = end - j0; if (m > 32) m = 32;
        int c = (j0 + lane < end) ? col[j0 + lane] : 0;
        for (int k = 0; k < m; k++) {
            int s = __shfl(c, k, 32);
            float2 v = f2[(size_t)s * 32 + lane];
            ax += v.x; ay += v.y;
        }
    }
    float di = dinv[node];
    float2 r; r.x = ax * di; r.y = ay * di;
    ((float2*)agg)[(size_t)node * 32 + lane] = r;
}

// ---------------- fused SAGE GEMM: out = agg@Wl + feat@Wr + b (WMMA bf16) ----------------
// K = 128 (agg | feat), 4 n-tiles of 16, 128 rows/block (16 per wave).
// A fragment = two contiguous 16B chunks of the lane's row (ds_load_b128 x2).
// B fragment = 32B per-lane load from pre-packed global buffer (L2-resident).

__global__ __launch_bounds__(256)
void k_sage_gemm(const float* __restrict__ agg, const float* __restrict__ feat,
                 const __bf16* __restrict__ Wp, const float* __restrict__ bias,
                 float* __restrict__ out, int n, int dout, int do_relu) {
    __shared__ __bf16 At[8][16][136];     // per-wave [row][K]; 272B row stride:
                                          // 16B-aligned + conflict-free bank rotation

    const int tid  = threadIdx.x;
    const int wave = tid >> 5, lane = tid & 31;
    const int half = lane >> 4, lr = lane & 15;

    // stage A tile: 16 rows x (64 agg | 64 feat) per wave, packed b64 stores
    const int r0 = blockIdx.x * 128 + wave * 16;
    #pragma unroll
    for (int it = 0; it < 8; it++) {
        int idx = it * 32 + lane;          // 0..255
        int row = idx >> 4;                // 0..15
        int q   = idx & 15;                // float4 slot
        int node = r0 + row; if (node >= n) node = n - 1;
        float4 va = ((const float4*)agg )[(size_t)node * 16 + q];
        float4 vf = ((const float4*)feat)[(size_t)node * 16 + q];
        int k = q * 4;
        v4bf pa = { (__bf16)va.x, (__bf16)va.y, (__bf16)va.z, (__bf16)va.w };
        v4bf pf = { (__bf16)vf.x, (__bf16)vf.y, (__bf16)vf.z, (__bf16)vf.w };
        *(v4bf*)(&At[wave][row][k])      = pa;
        *(v4bf*)(&At[wave][row][64 + k]) = pf;
    }
    __syncthreads();

    v8f acc[4];
    #pragma unroll
    for (int nt = 0; nt < 4; nt++)
        #pragma unroll
        for (int v = 0; v < 8; v++) acc[nt][v] = 0.f;

    const v16bf* wp = (const v16bf*)Wp;

    #pragma unroll
    for (int s = 0; s < 4; s++) {
        // A fragment (contiguous, matches packed B): i in [0,16) -> K = k0+16h+i
        union { v16bf v; v8bf h[2]; } ua;
        const __bf16* arow = &At[wave][lr][s * 32 + 16 * half];
        ua.h[0] = *(const v8bf*)(arow);
        ua.h[1] = *(const v8bf*)(arow + 8);
        #pragma unroll
        for (int nt = 0; nt < 4; nt++) {
            v16bf b = wp[(s * 4 + nt) * 32 + lane];
            acc[nt] = __builtin_amdgcn_wmma_f32_16x16x32_bf16(
                false, ua.v, false, b, (short)0, acc[nt], false, false);
        }
    }

    // bias + activation + store (C layout: VGPR v -> M = v + 8*half, N = lane%16)
    #pragma unroll
    for (int nt = 0; nt < 4; nt++) {
        int ncol = nt * 16 + lr;
        float bv = (ncol < dout) ? bias[ncol] : 0.f;
        #pragma unroll
        for (int v = 0; v < 8; v++) {
            int row = r0 + v + (half << 3);
            if (row < n && ncol < dout) {
                float val = acc[nt][v] + bv;
                if (do_relu) val = fmaxf(val, 0.f);
                out[(size_t)row * 64 + ncol] = val;   // stride 64 (padded)
            }
        }
    }
}

// ---------------- log_softmax over C classes (one wave per node) ----------------

__global__ __launch_bounds__(256)
void k_logsoftmax(const float* __restrict__ in, float* __restrict__ out, int n, int C) {
    int wave = threadIdx.x >> 5, lane = threadIdx.x & 31;
    int node = blockIdx.x * 8 + wave;
    if (node >= n) return;
    float v0 = (lane      < C) ? in[(size_t)node * 64 + lane]      : -__builtin_inff();
    float v1 = (lane + 32 < C) ? in[(size_t)node * 64 + lane + 32] : -__builtin_inff();
    float m = fmaxf(v0, v1);
    #pragma unroll
    for (int off = 16; off >= 1; off >>= 1) m = fmaxf(m, __shfl_xor(m, off, 32));
    float s = 0.f;
    if (lane      < C) s += __expf(v0 - m);
    if (lane + 32 < C) s += __expf(v1 - m);
    #pragma unroll
    for (int off = 16; off >= 1; off >>= 1) s += __shfl_xor(s, off, 32);
    float ls = __logf(s) + m;
    if (lane      < C) out[(size_t)node * C + lane]      = v0 - ls;
    if (lane + 32 < C) out[(size_t)node * C + lane + 32] = v1 - ls;
}

// ---------------- launch ----------------

extern "C" void kernel_launch(void* const* d_in, const int* in_sizes, int n_in,
                              void* d_out, int out_size, void* d_ws, size_t ws_size,
                              hipStream_t stream) {
    const float* x    = (const float*)d_in[0];
    const int*   edge = (const int*)  d_in[1];
    const float* Wl0  = (const float*)d_in[2];
    const float* Wr0  = (const float*)d_in[3];
    const float* b0   = (const float*)d_in[4];
    const float* Wl1  = (const float*)d_in[5];
    const float* Wr1  = (const float*)d_in[6];
    const float* b1   = (const float*)d_in[7];
    const float* Wl2  = (const float*)d_in[8];
    const float* Wr2  = (const float*)d_in[9];
    const float* b2   = (const float*)d_in[10];

    const int N = in_sizes[0] / 64;
    const int E = in_sizes[1] / 2;
    const int C = in_sizes[10];          // bias of last layer = NCLASS

    char* ws = (char*)d_ws;
    size_t off = 0;
    auto carve = [&](size_t bytes) -> void* {
        void* p = ws + off;
        off = (off + bytes + 255) & ~(size_t)255;
        return p;
    };
    int*    deg    = (int*)   carve((size_t)N * 4);
    int*    rowptr = (int*)   carve((size_t)(N + 1) * 4);
    int*    cnt    = (int*)   carve((size_t)N * 4);
    float*  dinv   = (float*) carve((size_t)N * 4);
    int*    col    = (int*)   carve((size_t)E * 4);
    __bf16* Wp0    = (__bf16*)carve((size_t)128 * 64 * 2);
    __bf16* Wp1    = (__bf16*)carve((size_t)128 * 64 * 2);
    __bf16* Wp2    = (__bf16*)carve((size_t)128 * 64 * 2);
    float*  aggb   = (float*) carve((size_t)N * 64 * 4);
    float*  hA     = (float*) carve((size_t)N * 64 * 4);
    float*  hB     = (float*) carve((size_t)N * 64 * 4);

    dim3 blk(256);

    // CSR build (once per call; counters zeroed for determinism)
    k_zero_i32<<<(N + 255) / 256, blk, 0, stream>>>(deg, N);
    k_zero_i32<<<(N + 255) / 256, blk, 0, stream>>>(cnt, N);
    k_degree  <<<(E + 255) / 256, blk, 0, stream>>>(edge, E, deg);
    k_scan    <<<1, 1024, 0, stream>>>(deg, rowptr, N);
    k_fill    <<<(E + 255) / 256, blk, 0, stream>>>(edge, E, rowptr, cnt, col);
    k_deginv  <<<(N + 255) / 256, blk, 0, stream>>>(rowptr, dinv, N);

    // pre-pack weight fragments (16 KB each, L2-resident broadcast)
    k_pack_w<<<32, blk, 0, stream>>>(Wl0, Wr0, 64, Wp0);
    k_pack_w<<<32, blk, 0, stream>>>(Wl1, Wr1, 64, Wp1);
    k_pack_w<<<32, blk, 0, stream>>>(Wl2, Wr2, C,  Wp2);

    const int gw = (N + 7) / 8;       // 8 waves per block, one wave per node
    const int gg = (N + 127) / 128;   // 128 rows per GEMM block

    // layer 0: x -> hA (relu)
    k_gather   <<<gw, blk, 0, stream>>>(x, rowptr, col, dinv, aggb, N);
    k_sage_gemm<<<gg, blk, 0, stream>>>(aggb, x, Wp0, b0, hA, N, 64, 1);
    // layer 1: hA -> hB (relu)
    k_gather   <<<gw, blk, 0, stream>>>(hA, rowptr, col, dinv, aggb, N);
    k_sage_gemm<<<gg, blk, 0, stream>>>(aggb, hA, Wp1, b1, hB, N, 64, 1);
    // layer 2: hB -> hA (no relu, dout = C)
    k_gather   <<<gw, blk, 0, stream>>>(hB, rowptr, col, dinv, aggb, N);
    k_sage_gemm<<<gg, blk, 0, stream>>>(aggb, hB, Wp2, b2, hA, N, C, 0);
    // log-softmax -> d_out
    k_logsoftmax<<<gw, blk, 0, stream>>>(hA, (float*)d_out, N, C);

    (void)n_in; (void)out_size; (void)ws_size;
}
